// MLALayer_17325898072449
// MI455X (gfx1250) — compile-verified, matrix operand order checked by reading, not run on hardware
//
#include <hip/hip_runtime.h>
#include <hip/hip_bf16.h>

typedef __attribute__((ext_vector_type(16))) _Float16 v16h;
typedef __attribute__((ext_vector_type(8)))  float    v8f;

constexpr int Bsz = 2, Tsz = 2048, Csz = 1024, Hh = 16, HSd = 64, KVR = 256, QR = 384;
constexpr int BT = Bsz * Tsz;

#ifndef __has_builtin
#define __has_builtin(x) 0
#endif

#if __has_builtin(__builtin_amdgcn_global_load_async_to_lds_b128)
#define ASYNC_LDS 1
#else
#define ASYNC_LDS 0
#endif

// Builtin prototype (from hipcc diagnostics): (int4 AS1*, int4 AS3*, imm, imm), non-const params.
typedef int v4i_vs __attribute__((vector_size(16)));
typedef __attribute__((address_space(1))) v4i_vs GV4;
typedef __attribute__((address_space(3))) v4i_vs LV4;

// 16-byte global -> LDS copy; async (ASYNCcnt-tracked) when available.
__device__ __forceinline__ void async_cp16(void* lds, const void* g) {
#if ASYNC_LDS
  __builtin_amdgcn_global_load_async_to_lds_b128((GV4*)g, (LV4*)lds, 0, 0);
#else
  *(uint4*)lds = *(const uint4*)g;
#endif
}

template <int N>
__device__ __forceinline__ void wait_async() {
#if ASYNC_LDS
#if __has_builtin(__builtin_amdgcn_s_wait_asynccnt)
  __builtin_amdgcn_s_wait_asynccnt(N);
#else
  asm volatile("s_wait_asynccnt %0" ::"i"(N) : "memory");
#endif
#endif
}

union FragU { v16h v; unsigned u[8]; _Float16 h[16]; };

// Gather a 16x32 f16 A-style fragment (B frags use [n][k]-transposed LDS tiles).
// Lanes 0-15: row=lane, K in {0..7,16..23}; lanes 16-31: K in {8..15,24..31}.
__device__ __forceinline__ v16h load_frag(const _Float16* lds, int rb, int ld, int kb, int ln) {
  const int row   = rb + (ln & 15);
  const int khalf = ((ln >> 4) & 1) * 8;
  const _Float16* p = lds + row * ld + kb + khalf;
  FragU f;
#pragma unroll
  for (int j = 0; j < 8; ++j) {
    const int K = ((j >> 2) << 4) + ((j & 3) << 1);
    f.u[j] = *(const unsigned*)(p + K);
  }
  return f.v;
}

__global__ __launch_bounds__(256) void f32_to_f16_kernel(const float* __restrict__ s,
                                                         _Float16* __restrict__ d, int n) {
  int i = blockIdx.x * 256 + threadIdx.x;
  if (i < n) d[i] = (_Float16)s[i];
}

// s is [K][N] f32, d is [N][K] f16 (pre-transposed weights -> GEMM B tiles are row-major copies)
__global__ __launch_bounds__(256) void f32_to_f16_t_kernel(const float* __restrict__ s,
                                                           _Float16* __restrict__ d,
                                                           int K, int N) {
  int i = blockIdx.x * 256 + threadIdx.x;
  if (i < K * N) {
    int n = i / K, k = i - n * K;
    d[i] = (_Float16)s[(size_t)k * N + n];
  }
}

// C[M,N] = A[M,K] @ Wt[N,K]^T. Block: 128x128 tile, 128 threads (4 waves, 64x64 each),
// K-step 32, double-buffered async LDS staging.
template <typename OutT>
__global__ __launch_bounds__(128) void gemm_wmma(const _Float16* __restrict__ A,
                                                 const _Float16* __restrict__ Wt,
                                                 OutT* __restrict__ Cout,
                                                 int M, int N, int K) {
  __shared__ _Float16 As[2][128 * 40];  // [row][k], ld=40
  __shared__ _Float16 Bs[2][128 * 40];  // [n][k],   ld=40
  const int tid = threadIdx.x, ln = tid & 31, wid = tid >> 5;
  const int mb = blockIdx.y * 128, nb = blockIdx.x * 128;
  const int wm = (wid >> 1) * 64, wn = (wid & 1) * 64;

  auto stage = [&](int buf, int kb) {  // 8 async b128 per thread (4 A + 4 B)
#pragma unroll
    for (int p = 0; p < 4; ++p) {
      const int idx = p * 128 + tid;
      const int r = idx >> 2, c = (idx & 3) * 8;  // 4 chunks per 32-half row
      async_cp16(&As[buf][r * 40 + c], A  + (size_t)(mb + r) * K + kb + c);
      async_cp16(&Bs[buf][r * 40 + c], Wt + (size_t)(nb + r) * K + kb + c);
    }
  };

  v8f acc[4][4] = {};
  const int nk = K / 32;
  stage(0, 0);
  for (int s = 0; s < nk; ++s) {
    const int buf = s & 1;
    if (s + 1 < nk) { stage(buf ^ 1, (s + 1) * 32); wait_async<8>(); }
    else            { wait_async<0>(); }
    __syncthreads();
    v16h bfrag[4];
#pragma unroll
    for (int j = 0; j < 4; ++j) bfrag[j] = load_frag(Bs[buf], wn + j * 16, 40, 0, ln);
#pragma unroll
    for (int i = 0; i < 4; ++i) {
      v16h a = load_frag(As[buf], wm + i * 16, 40, 0, ln);
#pragma unroll
      for (int j = 0; j < 4; ++j)
        acc[i][j] = __builtin_amdgcn_wmma_f32_16x16x32_f16(
            false, a, false, bfrag[j], (short)0, acc[i][j], false, false);
    }
    __syncthreads();  // all reads of buf done before it is restaged
  }
  const int col = ln & 15, mh = ((ln >> 4) & 1) * 8;
#pragma unroll
  for (int i = 0; i < 4; ++i)
#pragma unroll
    for (int j = 0; j < 4; ++j)
#pragma unroll
      for (int r = 0; r < 8; ++r) {
        const size_t m = (size_t)(mb + wm + i * 16 + mh + r);
        const size_t n = (size_t)(nb + wn + j * 16 + col);
        Cout[m * N + n] = (OutT)acc[i][j][r];
      }
}

// RoPE + pack q/k (d=128: content||rope) into [(b*H+h)*T + t] row-major layouts.
__global__ __launch_bounds__(256) void rope_pack_kernel(
    const _Float16* __restrict__ kvup, const _Float16* __restrict__ krl,
    const _Float16* __restrict__ qc,   const _Float16* __restrict__ qrl,
    const float* __restrict__ cosT,    const float* __restrict__ sinT,
    _Float16* __restrict__ qp, _Float16* __restrict__ kp) {
  unsigned i = blockIdx.x * 256 + threadIdx.x;  // (b, t, h, dd)
  const int dd = i & 63;
  const int h  = (i >> 6) & 15;
  const int t  = (i >> 10) & 2047;
  const int b  = i >> 21;
  const size_t bt  = (size_t)b * Tsz + t;
  const size_t src = bt * 1024 + h * 64 + dd;
  const float cv = cosT[t * 64 + dd], sv = sinT[t * 64 + dd];
  const size_t srcR = bt * 1024 + h * 64 + (dd < 32 ? dd + 32 : dd - 32);
  const float kr = (float)krl[src], qr = (float)qrl[src];
  const float krot = (dd < 32) ? -(float)krl[srcR] : (float)krl[srcR];
  const float qrot = (dd < 32) ? -(float)qrl[srcR] : (float)qrl[srcR];
  const size_t base = (size_t)(b * Hh + h) * Tsz + t;
  qp[base * 128 + dd]      = qc[src];
  qp[base * 128 + 64 + dd] = (_Float16)(qr * cv + qrot * sv);
  kp[base * 128 + dd]      = kvup[src];
  kp[base * 128 + 64 + dd] = (_Float16)(kr * cv + krot * sv);
}

// V transpose: kvup[b*T+t][h*64+hs] -> vp[(b*H+h)*64 + hs][T] (so attention V tiles are
// row-major [hs][key] copies). Grid: (T/64, B*H), 128 threads.
__global__ __launch_bounds__(128) void transpose_v_kernel(const _Float16* __restrict__ kvup,
                                                          _Float16* __restrict__ vp) {
  __shared__ _Float16 tile[64 * 72];
  const int tid = threadIdx.x;
  const int t0 = blockIdx.x * 64;
  const int bh = blockIdx.y, b = bh >> 4, h = bh & 15;
  {
    const int r = tid >> 1, c = (tid & 1) * 32;
#pragma unroll
    for (int i = 0; i < 4; ++i) {
      uint4 d = *(const uint4*)(kvup + ((size_t)(b * Tsz + t0 + r)) * 1024 + h * 64 + c + i * 8);
      *(uint4*)(tile + r * 72 + c + i * 8) = d;
    }
  }
  __syncthreads();
  {
    const int hs = tid >> 1, c = (tid & 1) * 32;
    union { uint4 q[4]; _Float16 hv[32]; } o;
#pragma unroll
    for (int i = 0; i < 32; ++i) o.hv[i] = tile[(c + i) * 72 + hs];
#pragma unroll
    for (int i = 0; i < 4; ++i)
      *(uint4*)(vp + ((size_t)(bh * 64 + hs)) * Tsz + t0 + c + i * 8) = o.q[i];
  }
}

// Flash attention: grid (T/64, B*H), 128 threads (4 waves), 64 query rows per block.
// K/V tiles double-buffered through the async LDS path.
__global__ __launch_bounds__(128) void attn_kernel(const _Float16* __restrict__ qp,
                                                   const _Float16* __restrict__ kp,
                                                   const _Float16* __restrict__ vp,
                                                   _Float16* __restrict__ yh) {
  __shared__ _Float16 Qs[64 * 136];        // [qrow][d]
  __shared__ _Float16 Ks[2][64 * 136];     // [key][d]   (== [N][K] for S = Q@K^T)
  __shared__ _Float16 Vs[2][64 * 72];      // [hs][key]  (== [N][K] for O = P@V)
  __shared__ _Float16 Ps[4 * 16 * 72];     // per-wave P 16x64
  const int tid = threadIdx.x, ln = tid & 31, w = tid >> 5;
  const int qt = blockIdx.x, bh = blockIdx.y;
  const int mh = ((ln >> 4) & 1) * 8, c0 = ln & 15;

  auto stage_kv = [&](int buf, int j) {  // 12 async b128 per thread (8 K + 4 V)
#pragma unroll
    for (int p = 0; p < 8; ++p) {        // K tile: 64 x 128 halfs
      const int idx = p * 128 + tid;
      const int r = idx >> 4, c = (idx & 15) * 8;
      async_cp16(&Ks[buf][r * 136 + c], kp + ((size_t)bh * Tsz + j * 64 + r) * 128 + c);
    }
#pragma unroll
    for (int p = 0; p < 4; ++p) {        // V tile: 64 x 64 halfs from transposed vp
      const int idx = p * 128 + tid;
      const int r = idx >> 3, c = (idx & 7) * 8;
      async_cp16(&Vs[buf][r * 72 + c], vp + ((size_t)(bh * 64 + r)) * Tsz + j * 64 + c);
    }
  };

  { // stage Q (8 async b128 per thread), overlapped with first K/V stage
#pragma unroll
    for (int p = 0; p < 8; ++p) {
      const int idx = p * 128 + tid;
      const int r = idx >> 4, c = (idx & 15) * 8;
      async_cp16(&Qs[r * 136 + c], qp + ((size_t)bh * Tsz + qt * 64 + r) * 128 + c);
    }
  }
  stage_kv(0, 0);

  v16h qa[4];
  v8f oacc[4] = {};
  float mrow[8], lrow[8];
#pragma unroll
  for (int r = 0; r < 8; ++r) { mrow[r] = -3.0e38f; lrow[r] = 0.f; }

  for (int j = 0; j <= qt; ++j) {
    const int cur = j & 1;
    if (j + 1 <= qt) { stage_kv(cur ^ 1, j + 1); wait_async<12>(); }
    else             { wait_async<0>(); }
    __syncthreads();
    if (j == 0) {
#pragma unroll
      for (int kk = 0; kk < 4; ++kk) qa[kk] = load_frag(Qs, w * 16, 136, kk * 32, ln);
    }

    // S = Q @ K^T  (16 q rows x 64 keys per wave)
    v8f sacc[4] = {};
#pragma unroll
    for (int n = 0; n < 4; ++n)
#pragma unroll
      for (int kk = 0; kk < 4; ++kk) {
        v16h b = load_frag(Ks[cur], n * 16, 136, kk * 32, ln);
        sacc[n] = __builtin_amdgcn_wmma_f32_16x16x32_f16(
            false, qa[kk], false, b, (short)0, sacc[n], false, false);
      }

    // scale + causal mask (diagonal tile only)
#pragma unroll
    for (int n = 0; n < 4; ++n)
#pragma unroll
      for (int r = 0; r < 8; ++r) {
        float s = sacc[n][r] * 0.125f;  // HS^-0.5
        if (j == qt) {
          const int kidx = j * 64 + n * 16 + c0;
          const int qidx = qt * 64 + w * 16 + mh + r;
          if (kidx > qidx) s = -1.0e30f;
        }
        sacc[n][r] = s;
      }

    // online softmax row max
    float mnew[8], alpha[8];
#pragma unroll
    for (int r = 0; r < 8; ++r) {
      float v = fmaxf(fmaxf(sacc[0][r], sacc[1][r]), fmaxf(sacc[2][r], sacc[3][r]));
#pragma unroll
      for (int off = 8; off >= 1; off >>= 1) v = fmaxf(v, __shfl_xor(v, off, 32));
      mnew[r]  = fmaxf(mrow[r], v);
      alpha[r] = __expf(mrow[r] - mnew[r]);
      mrow[r]  = mnew[r];
    }

    // P = exp(S - m), row sums, stash P in per-wave LDS (A-fragment source)
    float rs[8];
#pragma unroll
    for (int r = 0; r < 8; ++r) rs[r] = 0.f;
    _Float16* pw = Ps + w * 16 * 72;
#pragma unroll
    for (int n = 0; n < 4; ++n)
#pragma unroll
      for (int r = 0; r < 8; ++r) {
        float p = __expf(sacc[n][r] - mnew[r]);
        rs[r] += p;
        pw[(mh + r) * 72 + n * 16 + c0] = (_Float16)p;
      }
#pragma unroll
    for (int r = 0; r < 8; ++r) {
      float v = rs[r];
#pragma unroll
      for (int off = 8; off >= 1; off >>= 1) v += __shfl_xor(v, off, 32);
      lrow[r] = lrow[r] * alpha[r] + v;
    }
#pragma unroll
    for (int n = 0; n < 4; ++n)
#pragma unroll
      for (int r = 0; r < 8; ++r) oacc[n][r] *= alpha[r];

    // wave-local P write -> read ordering
    asm volatile("s_wait_dscnt 0x0" ::: "memory");

    // O += P @ V
#pragma unroll
    for (int kk = 0; kk < 2; ++kk) {
      v16h pa = load_frag(pw, 0, 72, kk * 32, ln);
#pragma unroll
      for (int n = 0; n < 4; ++n) {
        v16h bv = load_frag(Vs[cur], n * 16, 72, kk * 32, ln);
        oacc[n] = __builtin_amdgcn_wmma_f32_16x16x32_f16(
            false, pa, false, bv, (short)0, oacc[n], false, false);
      }
    }
    __syncthreads();  // all reads of cur tile done before it is restaged
  }

  // write y (f16, [b*T + t][h*64 + hs])
  const int b = bh >> 4, h = bh & 15;
#pragma unroll
  for (int n = 0; n < 4; ++n)
#pragma unroll
    for (int r = 0; r < 8; ++r) {
      const int t = qt * 64 + w * 16 + mh + r;
      yh[((size_t)(b * Tsz + t)) * 1024 + h * 64 + n * 16 + c0] =
          (_Float16)(oacc[n][r] / lrow[r]);
    }
}

extern "C" void kernel_launch(void* const* d_in, const int* in_sizes, int n_in,
                              void* d_out, int out_size, void* d_ws, size_t ws_size,
                              hipStream_t stream) {
  (void)in_sizes; (void)n_in; (void)out_size; (void)ws_size;
  const float* x    = (const float*)d_in[0];
  const float* cosT = (const float*)d_in[1];
  const float* sinT = (const float*)d_in[2];
  const float* Wkvd = (const float*)d_in[3];
  const float* Wkvu = (const float*)d_in[4];
  const float* Wkr  = (const float*)d_in[5];
  const float* Wqd  = (const float*)d_in[6];
  const float* Wqu  = (const float*)d_in[7];
  const float* Wqr  = (const float*)d_in[8];
  const float* Wout = (const float*)d_in[9];

  char* ws = (char*)d_ws;
  size_t off = 0;
  auto alloc = [&](size_t halves) -> _Float16* {
    _Float16* p = (_Float16*)(ws + off);
    off += (halves * sizeof(_Float16) + 255) & ~(size_t)255;
    return p;
  };
  _Float16* hx    = alloc((size_t)BT * Csz);
  _Float16* hWkvd = alloc((size_t)Csz * KVR);   // stored [N][K] (transposed)
  _Float16* hWkvu = alloc((size_t)KVR * Csz);
  _Float16* hWkr  = alloc((size_t)KVR * Csz);
  _Float16* hWqd  = alloc((size_t)Csz * QR);
  _Float16* hWqu  = alloc((size_t)QR * Csz);
  _Float16* hWqr  = alloc((size_t)QR * Csz);
  _Float16* hWout = alloc((size_t)Csz * Csz);
  _Float16* latkv = alloc((size_t)BT * KVR);
  _Float16* latq  = alloc((size_t)BT * QR);
  _Float16* kvup  = alloc((size_t)BT * Csz);
  _Float16* krl   = alloc((size_t)BT * Csz);
  _Float16* qc    = alloc((size_t)BT * Csz);
  _Float16* qrl   = alloc((size_t)BT * Csz);
  _Float16* qp    = alloc((size_t)Bsz * Hh * Tsz * 128);
  _Float16* kp    = alloc((size_t)Bsz * Hh * Tsz * 128);
  _Float16* vp    = alloc((size_t)Bsz * Hh * Tsz * 64);
  _Float16* yh    = alloc((size_t)BT * Csz);

  f32_to_f16_kernel<<<dim3((BT * Csz + 255) / 256), dim3(256), 0, stream>>>(x, hx, BT * Csz);
  auto convT = [&](const float* s, _Float16* d, int K, int N) {
    f32_to_f16_t_kernel<<<dim3((K * N + 255) / 256), dim3(256), 0, stream>>>(s, d, K, N);
  };
  convT(Wkvd, hWkvd, Csz, KVR);
  convT(Wkvu, hWkvu, KVR, Csz);
  convT(Wkr,  hWkr,  KVR, Csz);
  convT(Wqd,  hWqd,  Csz, QR);
  convT(Wqu,  hWqu,  QR,  Csz);
  convT(Wqr,  hWqr,  QR,  Csz);
  convT(Wout, hWout, Csz, Csz);

  // projections (weights pre-transposed to [N][K])
  gemm_wmma<_Float16><<<dim3(KVR / 128, BT / 128), 128, 0, stream>>>(hx, hWkvd, latkv, BT, KVR, Csz);
  gemm_wmma<_Float16><<<dim3(QR  / 128, BT / 128), 128, 0, stream>>>(hx, hWqd,  latq,  BT, QR,  Csz);
  gemm_wmma<_Float16><<<dim3(Csz / 128, BT / 128), 128, 0, stream>>>(latkv, hWkvu, kvup, BT, Csz, KVR);
  gemm_wmma<_Float16><<<dim3(Csz / 128, BT / 128), 128, 0, stream>>>(latkv, hWkr,  krl,  BT, Csz, KVR);
  gemm_wmma<_Float16><<<dim3(Csz / 128, BT / 128), 128, 0, stream>>>(latq,  hWqu,  qc,   BT, Csz, QR);
  gemm_wmma<_Float16><<<dim3(Csz / 128, BT / 128), 128, 0, stream>>>(latq,  hWqr,  qrl,  BT, Csz, QR);

  transpose_v_kernel<<<dim3(Tsz / 64, Bsz * Hh), dim3(128), 0, stream>>>(kvup, vp);
  rope_pack_kernel<<<dim3((BT * Hh * 64) / 256), dim3(256), 0, stream>>>(
      kvup, krl, qc, qrl, cosT, sinT, qp, kp);

  attn_kernel<<<dim3(Tsz / 64, Bsz * Hh), dim3(128), 0, stream>>>(qp, kp, vp, yh);

  gemm_wmma<float><<<dim3(Csz / 128, BT / 128), 128, 0, stream>>>(
      yh, hWout, (float*)d_out, BT, Csz, Hh * HSd);
}
